// WindowAttention_90271622628059
// MI455X (gfx1250) — matrix operand order repaired and checked by reading, not compiled, not run on hardware
//
#include <hip/hip_runtime.h>
#include <hip/hip_bf16.h>
#include <cstdint>
#include <cstddef>

#define DIMC   256
#define HEADS  8
#define NTOK   49
#define NWIN   4096
#define NWMASK 64
#define ROWS   (NWIN * NTOK)   /* 200704 = 128 * 1568 */
#define NN     (NTOK * NTOK)   /* 2401 */

typedef __attribute__((ext_vector_type(8)))  __bf16 bf16x8;
typedef __attribute__((ext_vector_type(16))) __bf16 bf16x16;
typedef __attribute__((ext_vector_type(8)))  float  f32x8;

static __device__ __forceinline__ bf16x16 frag_join(bf16x8 lo, bf16x8 hi) {
  return __builtin_shufflevector(lo, hi, 0,1,2,3,4,5,6,7,8,9,10,11,12,13,14,15);
}
static __device__ __forceinline__ bf16x8 load8(const __bf16* p) {
  return *reinterpret_cast<const bf16x8*>(p);
}
// A-matrix 16x32 bf16 fragment: lane holds row; low8 = K[8*hl..+7], high8 = K[8*hl+16..+7]
static __device__ __forceinline__ bf16x16 load_fragA(const __bf16* rowp, int hl) {
  const __bf16* p = rowp + hl * 8;
  return frag_join(load8(p), load8(p + 16));
}
// B-matrix 32x16 bf16 fragment: lane holds column n (rowp = &Bt[n][0]); low8 = K[8*hl..+7], high8 = K[8*hl+16..+7]
static __device__ __forceinline__ bf16x16 load_fragB(const __bf16* rowp, int hl) {
  const __bf16* p = rowp + hl * 8;
  return frag_join(load8(p), load8(p + 16));
}
static __device__ __forceinline__ f32x8 wmma_bf16(bf16x16 a, bf16x16 b, f32x8 c) {
  return __builtin_amdgcn_wmma_f32_16x16x32_bf16(false, a, false, b, (short)0, c, false, false);
}

// ---------------------------------------------------------------------------
// Kernel 0: weight transpose+convert to bf16, combined bias+mask table
// ---------------------------------------------------------------------------
__global__ void prep_kernel(const float* __restrict__ qkv_w,      // [256,768]
                            const float* __restrict__ proj_w,     // [256,256]
                            const float* __restrict__ bias_table, // [169,8]
                            const int*   __restrict__ rel_index,  // [49,49]
                            const float* __restrict__ mask,       // [64,49,49]
                            __bf16* __restrict__ w1t,   // [768,256]
                            __bf16* __restrict__ w2t,   // [256,256]
                            float*  __restrict__ bm)    // [64,8,49,49]
{
  const int T1 = 768 * 256;
  const int T2 = 256 * 256;
  const int T3 = NWMASK * HEADS * NN;
  int i = blockIdx.x * blockDim.x + threadIdx.x;
  if (i < T1) {
    int n = i >> 8, k = i & 255;
    w1t[n * 256 + k] = (__bf16)qkv_w[k * 768 + n];
  } else if (i < T1 + T2) {
    int j = i - T1;
    int n = j >> 8, k = j & 255;
    w2t[n * 256 + k] = (__bf16)proj_w[k * 256 + n];
  } else if (i < T1 + T2 + T3) {
    int j  = i - T1 - T2;
    int wh = j / NN;            // wm*8 + h
    int r  = j % NN;
    int wm = wh >> 3, h = wh & 7;
    bm[j] = bias_table[rel_index[r] * HEADS + h] + mask[wm * NN + r];
  }
}

// ---------------------------------------------------------------------------
// Kernel 1: qkv = x @ qkv_w + qkv_b, output bf16 [200704, 768]
// 128x128 C-tile per workgroup (8 waves); K in 32-chunks, ping-pong LDS with
// register prefetch of chunk k+1 overlapped with the WMMA block of chunk k.
// ---------------------------------------------------------------------------
__global__ __launch_bounds__(256) void qkv_gemm(const float* __restrict__ x,
                                                const __bf16* __restrict__ w1t,
                                                const float* __restrict__ qkv_b,
                                                __bf16* __restrict__ qkv_out)
{
  __shared__ __bf16 sA[2][128 * 40];   // stride 40 keeps 16B alignment, spreads banks
  __shared__ __bf16 sB[2][128 * 40];

  const int tid  = threadIdx.x;
  const int lane = tid & 31, wv = tid >> 5;
  const int hl   = lane >> 4, ln = lane & 15;
  const int Mbase = blockIdx.x * 128;
  const int Nbase = blockIdx.y * 128;

  f32x8 acc[8];
  const f32x8 zf = {0.f,0.f,0.f,0.f,0.f,0.f,0.f,0.f};
  #pragma unroll
  for (int i = 0; i < 8; ++i) acc[i] = zf;

  // per-thread slice: 4 x (16B of A, 8B of B) per chunk
  int rowA[4], colA[4];
  #pragma unroll
  for (int i = 0; i < 4; ++i) {
    int e = tid + i * 256;
    rowA[i] = e >> 3;
    colA[i] = (e & 7) << 2;
  }

  float4 ra[4];
  uint2  rb[4];
  // prefetch chunk 0
  #pragma unroll
  for (int i = 0; i < 4; ++i) {
    ra[i] = *reinterpret_cast<const float4*>(
        x + (size_t)(Mbase + rowA[i]) * DIMC + colA[i]);
    rb[i] = *reinterpret_cast<const uint2*>(
        w1t + (size_t)(Nbase + rowA[i]) * DIMC + colA[i]);
  }
  #pragma unroll
  for (int i = 0; i < 4; ++i) {
    __bf16* d = &sA[0][rowA[i] * 40 + colA[i]];
    d[0] = (__bf16)ra[i].x; d[1] = (__bf16)ra[i].y;
    d[2] = (__bf16)ra[i].z; d[3] = (__bf16)ra[i].w;
    *reinterpret_cast<uint2*>(&sB[0][rowA[i] * 40 + colA[i]]) = rb[i];
  }
  __syncthreads();

  for (int kc = 0; kc < 8; ++kc) {
    const int b = kc & 1;
    if (kc < 7) {                       // issue next chunk's global loads
      const int k0 = (kc + 1) * 32;
      #pragma unroll
      for (int i = 0; i < 4; ++i) {
        ra[i] = *reinterpret_cast<const float4*>(
            x + (size_t)(Mbase + rowA[i]) * DIMC + k0 + colA[i]);
        rb[i] = *reinterpret_cast<const uint2*>(
            w1t + (size_t)(Nbase + rowA[i]) * DIMC + k0 + colA[i]);
      }
    }
    // compute current chunk (loads above are only waited at the stores below)
    bf16x16 a = load_fragA(&sA[b][(wv * 16 + ln) * 40], hl);
    #pragma unroll
    for (int nt = 0; nt < 8; ++nt) {
      bf16x16 bt = load_fragB(&sB[b][(nt * 16 + ln) * 40], hl);
      acc[nt] = wmma_bf16(a, bt, acc[nt]);
    }
    if (kc < 7) {                       // fill the other buffer
      #pragma unroll
      for (int i = 0; i < 4; ++i) {
        __bf16* d = &sA[b ^ 1][rowA[i] * 40 + colA[i]];
        d[0] = (__bf16)ra[i].x; d[1] = (__bf16)ra[i].y;
        d[2] = (__bf16)ra[i].z; d[3] = (__bf16)ra[i].w;
        *reinterpret_cast<uint2*>(&sB[b ^ 1][rowA[i] * 40 + colA[i]]) = rb[i];
      }
      __syncthreads();
    }
  }

  #pragma unroll
  for (int nt = 0; nt < 8; ++nt) {
    int colg = Nbase + nt * 16 + ln;
    float bv = qkv_b[colg];
    #pragma unroll
    for (int r = 0; r < 8; ++r) {
      int rowg = Mbase + wv * 16 + hl * 8 + r;
      qkv_out[(size_t)rowg * 768 + colg] = (__bf16)(acc[nt][r] + bv);
    }
  }
}

// ---------------------------------------------------------------------------
// Kernel 2: fused attention + softmax + proj, one workgroup per window.
// 8 wave32s = 8 heads. Scores/P never touch HBM (LDS-resident, 192 KB/WG).
// ---------------------------------------------------------------------------
__global__ __launch_bounds__(256) void attn_proj(const __bf16* __restrict__ qkv,
                                                 const __bf16* __restrict__ w2t,
                                                 const float* __restrict__ bm,
                                                 const float* __restrict__ proj_b,
                                                 float* __restrict__ out)
{
  extern __shared__ __bf16 smem[];
  __bf16* sQ  = smem;            // 8 * 64*32
  __bf16* sK  = sQ  + 16384;     // 8 * 64*32
  __bf16* sVT = sK  + 16384;     // 8 * 32*64 (v transposed: [n][k])
  __bf16* sP  = sVT + 16384;     // 8 * 64*64
  __bf16* sO  = sP  + 32768;     // 64*256

  const int tid  = threadIdx.x;
  const int lane = tid & 31, h = tid >> 5;
  const int hl   = lane >> 4, ln = lane & 15;
  const int win  = blockIdx.x;
  const int rowbase = win * NTOK;
  const int wm   = win & (NWMASK - 1);

  __bf16* sQh  = sQ  + h * 2048;
  __bf16* sKh  = sK  + h * 2048;
  __bf16* sVTh = sVT + h * 2048;
  __bf16* sPh  = sP  + h * 4096;

  // ---- stage q/k/v for this head (16B vector loads), zero-pad to 64 tokens -
  const __bf16 z = (__bf16)0.f;
  const bf16x8 z8 = {z,z,z,z,z,z,z,z};
  #pragma unroll
  for (int it = 0; it < 8; ++it) {
    int e  = it * 32 + lane;      // 0..255: 64 rows x 4 chunks of 8 bf16
    int m  = e >> 2;
    int c8 = (e & 3) << 3;
    bf16x8 qv = z8, kv = z8, vv = z8;
    if (m < NTOK) {
      const __bf16* rp = qkv + (size_t)(rowbase + m) * 768 + h * 32 + c8;
      qv = load8(rp); kv = load8(rp + 256); vv = load8(rp + 512);
    }
    *reinterpret_cast<bf16x8*>(sQh + m * 32 + c8) = qv;
    *reinterpret_cast<bf16x8*>(sKh + m * 32 + c8) = kv;
    #pragma unroll
    for (int j = 0; j < 8; ++j) sVTh[(c8 + j) * 64 + m] = vv[j];  // transpose
  }
  __syncthreads();

  // ---- scores = q @ k^T (64x64 padded, K=32: one WMMA step) ----------------
  const f32x8 zf = {0.f,0.f,0.f,0.f,0.f,0.f,0.f,0.f};
  f32x8 accS[4][4];
  #pragma unroll
  for (int mt = 0; mt < 4; ++mt)
    #pragma unroll
    for (int nt = 0; nt < 4; ++nt) accS[mt][nt] = zf;

  bf16x16 aq[4];
  #pragma unroll
  for (int mt = 0; mt < 4; ++mt)
    aq[mt] = load_fragA(sQh + (mt * 16 + ln) * 32, hl);
  #pragma unroll
  for (int nt = 0; nt < 4; ++nt) {
    bf16x16 bk = load_fragB(sKh + (nt * 16 + ln) * 32, hl);
    #pragma unroll
    for (int mt = 0; mt < 4; ++mt)
      accS[mt][nt] = wmma_bf16(aq[mt], bk, accS[mt][nt]);
  }

  // ---- softmax (scale + combined bias+mask), shuffle row reductions --------
  const float SCALE = 0.17677669529663687f;   // 1/sqrt(32)
  const float* bmW = bm + (size_t)(wm * HEADS + h) * NN;
  #pragma unroll
  for (int mt = 0; mt < 4; ++mt) {
    #pragma unroll
    for (int r = 0; r < 8; ++r) {
      int m = mt * 16 + hl * 8 + r;
      float sv[4];
      #pragma unroll
      for (int nt = 0; nt < 4; ++nt) {
        int n = nt * 16 + ln;
        float s = -1e30f;                     // finite mask: padded rows stay NaN-free
        if (m < NTOK && n < NTOK)
          s = accS[mt][nt][r] * SCALE + bmW[m * NTOK + n];
        sv[nt] = s;
      }
      float mx = fmaxf(fmaxf(sv[0], sv[1]), fmaxf(sv[2], sv[3]));
      #pragma unroll
      for (int d = 8; d >= 1; d >>= 1) mx = fmaxf(mx, __shfl_xor(mx, d, 32));
      float es[4], sum = 0.f;
      #pragma unroll
      for (int nt = 0; nt < 4; ++nt) { es[nt] = __expf(sv[nt] - mx); sum += es[nt]; }
      #pragma unroll
      for (int d = 8; d >= 1; d >>= 1) sum += __shfl_xor(sum, d, 32);
      float inv = 1.f / sum;
      #pragma unroll
      for (int nt = 0; nt < 4; ++nt)
        sPh[m * 64 + nt * 16 + ln] = (__bf16)(es[nt] * inv);
    }
  }
  __syncthreads();

  // ---- O = P @ V  (M=64pad, N=32, K=64pad) ---------------------------------
  f32x8 accO[4][2];
  #pragma unroll
  for (int mt = 0; mt < 4; ++mt)
    #pragma unroll
    for (int nt = 0; nt < 2; ++nt) accO[mt][nt] = zf;
  #pragma unroll
  for (int kc = 0; kc < 2; ++kc) {
    bf16x16 ap[4];
    #pragma unroll
    for (int mt = 0; mt < 4; ++mt)
      ap[mt] = load_fragA(sPh + (mt * 16 + ln) * 64 + kc * 32, hl);
    #pragma unroll
    for (int nt = 0; nt < 2; ++nt) {
      bf16x16 bv = load_fragB(sVTh + (nt * 16 + ln) * 64 + kc * 32, hl);
      #pragma unroll
      for (int mt = 0; mt < 4; ++mt)
        accO[mt][nt] = wmma_bf16(ap[mt], bv, accO[mt][nt]);
    }
  }
  // write this head's 32 output columns into the shared [64,256] tile
  #pragma unroll
  for (int mt = 0; mt < 4; ++mt)
    #pragma unroll
    for (int nt = 0; nt < 2; ++nt) {
      int col = h * 32 + nt * 16 + ln;
      #pragma unroll
      for (int r = 0; r < 8; ++r) {
        int m = mt * 16 + hl * 8 + r;
        sO[m * 256 + col] = (__bf16)(accO[mt][nt][r]);
      }
    }
  __syncthreads();

  // ---- fused proj: out = O @ proj_w + proj_b; wave h -> 32 output cols -----
  f32x8 accP[4][2];
  #pragma unroll
  for (int mt = 0; mt < 4; ++mt)
    #pragma unroll
    for (int nt = 0; nt < 2; ++nt) accP[mt][nt] = zf;
  for (int kc = 0; kc < 8; ++kc) {
    bf16x16 ao[4];
    #pragma unroll
    for (int mt = 0; mt < 4; ++mt)
      ao[mt] = load_fragA(sO + (mt * 16 + ln) * 256 + kc * 32, hl);
    #pragma unroll
    for (int nt = 0; nt < 2; ++nt) {
      bf16x16 bw = load_fragB(w2t + (size_t)(h * 32 + nt * 16 + ln) * 256 + kc * 32, hl);
      #pragma unroll
      for (int mt = 0; mt < 4; ++mt)
        accP[mt][nt] = wmma_bf16(ao[mt], bw, accP[mt][nt]);
    }
  }
  #pragma unroll
  for (int nt = 0; nt < 2; ++nt) {
    int colg = h * 32 + nt * 16 + ln;
    float bv = proj_b[colg];
    #pragma unroll
    for (int mt = 0; mt < 4; ++mt)
      #pragma unroll
      for (int r = 0; r < 8; ++r) {
        int m = mt * 16 + hl * 8 + r;
        if (m < NTOK)
          out[(size_t)(rowbase + m) * DIMC + colg] = accP[mt][nt][r] + bv;
      }
  }
}

// ---------------------------------------------------------------------------
extern "C" void kernel_launch(void* const* d_in, const int* in_sizes, int n_in,
                              void* d_out, int out_size, void* d_ws, size_t ws_size,
                              hipStream_t stream) {
  (void)in_sizes; (void)n_in; (void)out_size; (void)ws_size;
  const float* x          = (const float*)d_in[0];
  const float* mask       = (const float*)d_in[1];
  const float* qkv_w      = (const float*)d_in[2];
  const float* qkv_b      = (const float*)d_in[3];
  const float* proj_w     = (const float*)d_in[4];
  const float* proj_b     = (const float*)d_in[5];
  const float* bias_table = (const float*)d_in[6];
  const int*   rel_index  = (const int*)d_in[7];
  float* out = (float*)d_out;

  // workspace layout (bytes)
  char* ws = (char*)d_ws;
  const size_t QKV_BYTES = (size_t)ROWS * 768 * 2;       // 308,281,344 (bf16)
  const size_t W1T_BYTES = (size_t)768 * 256 * 2;
  const size_t W2T_BYTES = (size_t)256 * 256 * 2;
  __bf16* qkv_buf = (__bf16*)(ws);
  __bf16* w1t     = (__bf16*)(ws + QKV_BYTES);
  __bf16* w2t     = (__bf16*)(ws + QKV_BYTES + W1T_BYTES);
  float*  bm      = (float*) (ws + QKV_BYTES + W1T_BYTES + W2T_BYTES); // [64,8,49,49]

  // prep: weight transpose/convert + combined bias+mask table
  {
    const int T = 768*256 + 256*256 + NWMASK*HEADS*NN;
    prep_kernel<<<(T + 255) / 256, 256, 0, stream>>>(
        qkv_w, proj_w, bias_table, rel_index, mask, w1t, w2t, bm);
  }
  // qkv GEMM: 1568 x 6 tiles of 128x128
  qkv_gemm<<<dim3(ROWS / 128, 768 / 128), 256, 0, stream>>>(x, w1t, qkv_b, qkv_buf);

  // fused attention + proj: one WG per window, 192 KB dynamic LDS
  const int SMEM = 196608;
  (void)hipFuncSetAttribute((const void*)attn_proj,
                            hipFuncAttributeMaxDynamicSharedMemorySize, SMEM);
  attn_proj<<<NWIN, 256, SMEM, stream>>>(qkv_buf, w2t, bm, proj_b, out);
}
